// Mamba_37744172597590
// MI455X (gfx1250) — matrix operand order repaired and checked by reading, not compiled, not run on hardware
//
#include <hip/hip_runtime.h>
#include <hip/hip_bf16.h>
#include <math.h>

// ---------------------------------------------------------------------------
// Mamba (3 layers) for MI455X / gfx1250.
//  - All GEMMs via v_wmma_f32_16x16x32_f16 (f16 in, f32 accumulate).
//  - GEMM register-blocked: one wave owns a 16x64 strip (1 A fragment reused
//    against 4 B fragments, 4 f32 accumulators) to raise WMMA:VMEM ratio.
//  - Selective scan parallelized: one lane per (batch,channel,state), 16-lane
//    shuffle reduction for the state contraction (wave32).
// ---------------------------------------------------------------------------

#define D_MODEL 512
#define D_INNER 1024
#define DT_RANK 32
#define D_STATE 16
#define D_CONV  4
#define BATCH   2
#define SEQ     1024
#define NTOK    (BATCH * SEQ)       // 2048 tokens
#define N_LAYERS 3
#define RMS_EPS 1e-5f

typedef __attribute__((ext_vector_type(16))) _Float16 v16h;
typedef __attribute__((ext_vector_type(8)))  _Float16 v8h;
typedef __attribute__((ext_vector_type(8)))  float    v8f;

static __device__ __forceinline__ v16h load_frag16(const _Float16* p, int off0, int off1)
{
    v8h x0 = *(const v8h*)(p + off0);
    v8h x1 = *(const v8h*)(p + off1);
    return __builtin_shufflevector(x0, x1, 0,1,2,3,4,5,6,7, 8,9,10,11,12,13,14,15);
}

// ---------------------------------------------------------------------------
// WMMA GEMM:  C[M][N] = A[M][K] * W[N][K]^T  (+ optional residual)
// A, W row-major f16 with K contiguous; C f32 row-major.
// One wave computes a 16x64 strip (4 WMMA tiles sharing one A fragment).
// M multiple of 16, N multiple of 64, K multiple of 32.
// ---------------------------------------------------------------------------
__global__ void wmma_gemm_nt(const _Float16* __restrict__ A,
                             const _Float16* __restrict__ W,
                             const float*    __restrict__ residual,
                             float*          __restrict__ C,
                             int M, int N, int K)
{
    const int wave   = (int)((blockIdx.x * blockDim.x + threadIdx.x) >> 5);
    const int lane   = (int)(threadIdx.x & 31u);
    const int strips = N >> 6;                 // 64-wide supertiles
    const int tm  = wave / strips;
    const int tn4 = wave % strips;
    if (tm * 16 >= M) return;                  // wave-uniform guard (EXEC all-1)

    const int row = lane & 15;                 // M-row (A) / N-col (B) in tile
    const int hi  = lane >> 4;                 // lane half selector

    const _Float16* __restrict__ arow = A + (size_t)(tm * 16 + row) * K;
    const _Float16* __restrict__ w0 = W + (size_t)(tn4 * 64 +  0 + row) * K;
    const _Float16* __restrict__ w1 = W + (size_t)(tn4 * 64 + 16 + row) * K;
    const _Float16* __restrict__ w2 = W + (size_t)(tn4 * 64 + 32 + row) * K;
    const _Float16* __restrict__ w3 = W + (size_t)(tn4 * 64 + 48 + row) * K;

    v8f acc0 = {}, acc1 = {}, acc2 = {}, acc3 = {};
    for (int k = 0; k < K; k += 32) {
        // A fragment: lane holds row, K = kbase..kbase+15 contiguous
        const int ka = k + hi * 16;
        v16h a = load_frag16(arow, ka, ka + 8);
        // B fragments: lane holds column (row), K = kbase..kbase+15 contiguous
        const int kb = k + hi * 16;
        v16h b0 = load_frag16(w0, kb, kb + 8);
        v16h b1 = load_frag16(w1, kb, kb + 8);
        v16h b2 = load_frag16(w2, kb, kb + 8);
        v16h b3 = load_frag16(w3, kb, kb + 8);
        if (k + 32 < K) {                      // hint next K-chunk toward caches
            __builtin_prefetch(arow + k + 32, 0, 1);
            __builtin_prefetch(w0 + k + 32, 0, 1);
            __builtin_prefetch(w2 + k + 32, 0, 1);
        }
        acc0 = __builtin_amdgcn_wmma_f32_16x16x32_f16(false, a, false, b0, (short)0, acc0, false, false);
        acc1 = __builtin_amdgcn_wmma_f32_16x16x32_f16(false, a, false, b1, (short)0, acc1, false, false);
        acc2 = __builtin_amdgcn_wmma_f32_16x16x32_f16(false, a, false, b2, (short)0, acc2, false, false);
        acc3 = __builtin_amdgcn_wmma_f32_16x16x32_f16(false, a, false, b3, (short)0, acc3, false, false);
    }

    // D layout: VGPR i -> M = i + 8*hi (lanes 0-15 / 16-31), N = lane&15
    const int m0 = tm * 16 + hi * 8;
    const v8f accs[4] = { acc0, acc1, acc2, acc3 };
    #pragma unroll
    for (int j = 0; j < 4; ++j) {
        const int n = tn4 * 64 + j * 16 + row;
        #pragma unroll
        for (int i = 0; i < 8; ++i) {
            size_t idx = (size_t)(m0 + i) * N + n;
            float v = accs[j][i];
            if (residual) v += residual[idx];
            C[idx] = v;
        }
    }
}

// ---------------------------------------------------------------------------
// RMSNorm + cast to f16. One block (256 thr) per token, D = 512.
// ---------------------------------------------------------------------------
__global__ void rmsnorm_cast(const float* __restrict__ x,
                             const float* __restrict__ w,
                             _Float16*    __restrict__ u)
{
    const int t = blockIdx.x;
    const float* xr = x + (size_t)t * D_MODEL;
    float ss = 0.f;
    for (int i = threadIdx.x; i < D_MODEL; i += blockDim.x) {
        float v = xr[i];
        ss += v * v;
    }
    __shared__ float red[256];
    red[threadIdx.x] = ss;
    __syncthreads();
    for (int s = 128; s > 0; s >>= 1) {
        if ((int)threadIdx.x < s) red[threadIdx.x] += red[threadIdx.x + s];
        __syncthreads();
    }
    const float scale = rsqrtf(red[0] / (float)D_MODEL + RMS_EPS);
    for (int i = threadIdx.x; i < D_MODEL; i += blockDim.x)
        u[(size_t)t * D_MODEL + i] = (_Float16)(xr[i] * scale * w[i]);
}

// ---------------------------------------------------------------------------
// f32 -> f16 cast (grid-stride), used for weight conversion.
// ---------------------------------------------------------------------------
__global__ void cast_f32_f16(const float* __restrict__ in,
                             _Float16*    __restrict__ out, int n)
{
    for (int i = blockIdx.x * blockDim.x + threadIdx.x; i < n;
         i += blockDim.x * gridDim.x)
        out[i] = (_Float16)in[i];
}

// ---------------------------------------------------------------------------
// Causal depthwise conv (window 4) + bias + SiLU over x-half of xz.
// Emits both f16 (GEMM operand) and f32 (scan operand) copies.
// ---------------------------------------------------------------------------
__global__ void conv_silu(const float* __restrict__ xz,     // (NTOK, 2*D_INNER)
                          const float* __restrict__ cw,     // (D_INNER, 4)
                          const float* __restrict__ cb,     // (D_INNER)
                          _Float16*    __restrict__ xc16,   // (NTOK, D_INNER)
                          float*       __restrict__ xc32)
{
    const int i = blockIdx.x * blockDim.x + threadIdx.x;
    if (i >= NTOK * D_INNER) return;
    const int e = i % D_INNER;
    const int t = (i / D_INNER) % SEQ;
    const int b = i / (D_INNER * SEQ);
    float acc = cb[e];
    #pragma unroll
    for (int j = 0; j < D_CONV; ++j) {
        const int tt = t - (D_CONV - 1) + j;
        if (tt >= 0)
            acc += cw[e * D_CONV + j] *
                   xz[((size_t)(b * SEQ + tt)) * (2 * D_INNER) + e];
    }
    const float s = acc / (1.f + __expf(-acc));   // SiLU
    xc32[i] = s;
    xc16[i] = (_Float16)s;
}

// ---------------------------------------------------------------------------
// Extract first DT_RANK columns of dbc as f16 (dt_proj GEMM operand).
// ---------------------------------------------------------------------------
__global__ void extract_delta(const float* __restrict__ dbc,   // (NTOK, 64)
                              _Float16*    __restrict__ out)   // (NTOK, 32)
{
    const int i = blockIdx.x * blockDim.x + threadIdx.x;
    if (i >= NTOK * DT_RANK) return;
    const int t = i >> 5;
    const int c = i & 31;
    out[i] = (_Float16)dbc[(size_t)t * (DT_RANK + 2 * D_STATE) + c];
}

// ---------------------------------------------------------------------------
// Selective scan. One lane per (b, e, n): 2048*16 = 32768 lanes.
// 16-lane shuffle butterfly reduces over the state dim each step.
// ---------------------------------------------------------------------------
__global__ void selective_scan(const float* __restrict__ dbc,    // (NTOK,64)
                               const float* __restrict__ dpre,   // (NTOK,D_INNER)
                               const float* __restrict__ dtb,    // (D_INNER)
                               const float* __restrict__ A_log,  // (D_INNER,16)
                               const float* __restrict__ Dskip,  // (D_INNER)
                               const float* __restrict__ xc32,   // (NTOK,D_INNER)
                               const float* __restrict__ xz,     // (NTOK,2*D_INNER)
                               _Float16*    __restrict__ yout)   // (NTOK,D_INNER)
{
    const int gid = blockIdx.x * blockDim.x + threadIdx.x;
    const int n  = gid & (D_STATE - 1);
    const int ch = gid >> 4;                 // 0 .. BATCH*D_INNER-1
    if (ch >= BATCH * D_INNER) return;
    const int e = ch % D_INNER;
    const int b = ch / D_INNER;

    const float A    = -__expf(A_log[e * D_STATE + n]);
    const float Dv   = Dskip[e];
    const float dtbv = dtb[e];

    float h = 0.f;
    const size_t base = (size_t)b * SEQ;
    for (int t = 0; t < SEQ; ++t) {
        const size_t tok = base + t;
        float dp = dpre[tok * D_INNER + e] + dtbv;
        float delta = (dp > 20.f) ? dp : log1pf(__expf(dp));      // softplus
        const float xv = xc32[tok * D_INNER + e];
        const float Bm = dbc[tok * 64 + DT_RANK + n];
        const float Cm = dbc[tok * 64 + DT_RANK + D_STATE + n];
        h = __expf(delta * A) * h + delta * xv * Bm;
        float y = h * Cm;
        // reduce y across the 16 state lanes
        #pragma unroll
        for (int off = 8; off > 0; off >>= 1)
            y += __shfl_xor(y, off, 16);
        if (n == 0) {
            const float zv = xz[tok * (2 * D_INNER) + D_INNER + e];
            const float g  = zv / (1.f + __expf(-zv));            // SiLU(z)
            yout[tok * D_INNER + e] = (_Float16)((y + Dv * xv) * g);
        }
    }
}

// ---------------------------------------------------------------------------
// Host launcher
// ---------------------------------------------------------------------------
extern "C" void kernel_launch(void* const* d_in, const int* in_sizes, int n_in,
                              void* d_out, int out_size, void* d_ws, size_t ws_size,
                              hipStream_t stream)
{
    const float* x_in   = (const float*)d_in[0];   // (B,L,512)
    const float* ipw    = (const float*)d_in[1];   // (3,2048,512)
    const float* convw  = (const float*)d_in[2];   // (3,1024,1,4)
    const float* convb  = (const float*)d_in[3];   // (3,1024)
    const float* xpw    = (const float*)d_in[4];   // (3,64,1024)
    const float* dtw    = (const float*)d_in[5];   // (3,1024,32)
    const float* dtb    = (const float*)d_in[6];   // (3,1024)
    const float* A_log  = (const float*)d_in[7];   // (3,1024,16)
    const float* Dskip  = (const float*)d_in[8];   // (3,1024)
    const float* opw    = (const float*)d_in[9];   // (3,512,1024)
    const float* normw  = (const float*)d_in[10];  // (3,512)

    // ---- workspace layout ----
    size_t off = 0;
    auto alloc = [&](size_t bytes) -> void* {
        void* p = (char*)d_ws + off;
        off += (bytes + 255) & ~(size_t)255;
        return p;
    };
    float*     x_cur   = (float*)    alloc((size_t)NTOK * D_MODEL * 4);
    _Float16*  u16     = (_Float16*) alloc((size_t)NTOK * D_MODEL * 2);
    _Float16*  w16_in  = (_Float16*) alloc((size_t)2 * D_INNER * D_MODEL * 2);
    _Float16*  w16_xp  = (_Float16*) alloc((size_t)64 * D_INNER * 2);
    _Float16*  w16_dt  = (_Float16*) alloc((size_t)D_INNER * DT_RANK * 2);
    _Float16*  w16_out = (_Float16*) alloc((size_t)D_MODEL * D_INNER * 2);
    float*     xz      = (float*)    alloc((size_t)NTOK * 2 * D_INNER * 4);
    _Float16*  xc16    = (_Float16*) alloc((size_t)NTOK * D_INNER * 2);
    float*     xc32    = (float*)    alloc((size_t)NTOK * D_INNER * 4);
    float*     dbc     = (float*)    alloc((size_t)NTOK * 64 * 4);
    _Float16*  delta16 = (_Float16*) alloc((size_t)NTOK * DT_RANK * 2);
    float*     dpre    = (float*)    alloc((size_t)NTOK * D_INNER * 4);
    _Float16*  y16     = (_Float16*) alloc((size_t)NTOK * D_INNER * 2);
    (void)ws_size;

    // residual state <- input
    hipMemcpyAsync(x_cur, x_in, (size_t)NTOK * D_MODEL * sizeof(float),
                   hipMemcpyDeviceToDevice, stream);

    auto cast = [&](const float* src, _Float16* dst, int n) {
        int blocks = (n + 255) / 256;
        if (blocks > 2048) blocks = 2048;
        cast_f32_f16<<<blocks, 256, 0, stream>>>(src, dst, n);
    };
    auto gemm = [&](const _Float16* A, const _Float16* W, const float* res,
                    float* C, int M, int N, int K) {
        int strips = (M / 16) * (N / 64);     // one 16x64 strip per wave
        int grid   = (strips + 7) / 8;        // 8 waves (256 threads) per block
        wmma_gemm_nt<<<grid, 256, 0, stream>>>(A, W, res, C, M, N, K);
    };

    for (int l = 0; l < N_LAYERS; ++l) {
        // weight casts
        cast(ipw + (size_t)l * 2 * D_INNER * D_MODEL, w16_in, 2 * D_INNER * D_MODEL);
        cast(xpw + (size_t)l * 64 * D_INNER,          w16_xp, 64 * D_INNER);
        cast(dtw + (size_t)l * D_INNER * DT_RANK,     w16_dt, D_INNER * DT_RANK);
        cast(opw + (size_t)l * D_MODEL * D_INNER,     w16_out, D_MODEL * D_INNER);

        // RMSNorm -> u16
        rmsnorm_cast<<<NTOK, 256, 0, stream>>>(x_cur, normw + (size_t)l * D_MODEL, u16);

        // in_proj: xz = u @ Win^T   (2048 x 2048, K=512)
        gemm(u16, w16_in, nullptr, xz, NTOK, 2 * D_INNER, D_MODEL);

        // depthwise causal conv + SiLU
        conv_silu<<<(NTOK * D_INNER + 255) / 256, 256, 0, stream>>>(
            xz, convw + (size_t)l * D_INNER * D_CONV,
            convb + (size_t)l * D_INNER, xc16, xc32);

        // x_proj: dbc = xconv @ Wxp^T   (2048 x 64, K=1024)
        gemm(xc16, w16_xp, nullptr, dbc, NTOK, 64, D_INNER);

        // delta rank columns -> f16
        extract_delta<<<(NTOK * DT_RANK + 255) / 256, 256, 0, stream>>>(dbc, delta16);

        // dt_proj: dpre = delta @ Wdt^T   (2048 x 1024, K=32)
        gemm(delta16, w16_dt, nullptr, dpre, NTOK, D_INNER, DT_RANK);

        // selective scan + gate -> y16
        selective_scan<<<(BATCH * D_INNER * D_STATE) / 256, 256, 0, stream>>>(
            dbc, dpre, dtb + (size_t)l * D_INNER,
            A_log + (size_t)l * D_INNER * D_STATE,
            Dskip + (size_t)l * D_INNER, xc32, xz, y16);

        // out_proj + residual: x_cur = y @ Wout^T + x_cur   (2048 x 512, K=1024)
        gemm(y16, w16_out, x_cur, x_cur, NTOK, D_MODEL, D_INNER);
    }

    hipMemcpyAsync(d_out, x_cur, (size_t)out_size * sizeof(float),
                   hipMemcpyDeviceToDevice, stream);
}
